// AGIFORMERBlock_67499706024580
// MI455X (gfx1250) — matrix operand order repaired and checked by reading, not compiled, not run on hardware
//
#include <hip/hip_runtime.h>
#include <stdint.h>

// ---------------------------------------------------------------- constants
static constexpr int Bq = 2, Sq = 2048, Dq = 1024, Hh = 16, DH = 64;
static constexpr int Fq = 4096, Eq = 8;
static constexpr int NT = Bq * Sq;   // 4096 tokens
static constexpr int NA = 2 * NT;    // 8192 routed (token,expert) rows

// ---------------------------------------------------------------- types
typedef __attribute__((ext_vector_type(16))) __bf16 v16bf;
typedef __attribute__((ext_vector_type(8)))  float  v8f;
typedef __attribute__((ext_vector_type(4)))  int    v4i;
typedef __attribute__((address_space(1))) v4i gv4i;   // global int4
typedef __attribute__((address_space(3))) v4i lv4i;   // LDS int4

#if defined(__gfx1250__) && __has_builtin(__builtin_amdgcn_global_load_async_to_lds_b128)
#define HAVE_ASYNC_LDS 1
#else
#define HAVE_ASYNC_LDS 0
#endif

__device__ __forceinline__ unsigned short f2bf(float f) {
  unsigned int u = __float_as_uint(f);
  u += 0x7fffu + ((u >> 16) & 1u);        // round-to-nearest-even
  return (unsigned short)(u >> 16);
}
__device__ __forceinline__ unsigned int pack2(float a, float b) {
  return (unsigned int)f2bf(a) | ((unsigned int)f2bf(b) << 16);
}

// 16-byte global->LDS copy: async DMA (ASYNCcnt) when available.
__device__ __forceinline__ void copy16(unsigned short* l, const unsigned short* g) {
#if HAVE_ASYNC_LDS
  __builtin_amdgcn_global_load_async_to_lds_b128((gv4i*)g, (lv4i*)l, 0, 0);
#else
  *(uint4*)l = *(const uint4*)g;
#endif
}
__device__ __forceinline__ void wait_async() {
#if HAVE_ASYNC_LDS
#if __has_builtin(__builtin_amdgcn_s_wait_asynccnt)
  __builtin_amdgcn_s_wait_asynccnt(0);
#else
  asm volatile("s_wait_asynccnt 0x0" ::: "memory");
#endif
#endif
}

// Build a 16-element bf16 fragment from two 16-byte contiguous LDS chunks.
__device__ __forceinline__ v16bf frag_from(const unsigned short* p0,
                                           const unsigned short* p1) {
  union { struct { uint4 a, b; } u; v16bf v; } t;
  t.u.a = *(const uint4*)p0;
  t.u.b = *(const uint4*)p1;
  return t.v;
}

__device__ __forceinline__ v8f wmma_bf16(v16bf a, v16bf b, v8f c) {
  return __builtin_amdgcn_wmma_f32_16x16x32_bf16(
      false, a, false, b, (short)0, c, false, false);
}

// Pure-VALU DPP16 butterfly reductions across a 16-lane half.
__device__ __forceinline__ float dpp_max16(float v) {
  v = fmaxf(v, __builtin_bit_cast(float, __builtin_amdgcn_update_dpp(
      0, __builtin_bit_cast(int, v), 0xB1, 0xF, 0xF, true)));
  v = fmaxf(v, __builtin_bit_cast(float, __builtin_amdgcn_update_dpp(
      0, __builtin_bit_cast(int, v), 0x4E, 0xF, 0xF, true)));
  v = fmaxf(v, __builtin_bit_cast(float, __builtin_amdgcn_update_dpp(
      0, __builtin_bit_cast(int, v), 0x141, 0xF, 0xF, true)));
  v = fmaxf(v, __builtin_bit_cast(float, __builtin_amdgcn_update_dpp(
      0, __builtin_bit_cast(int, v), 0x140, 0xF, 0xF, true)));
  return v;
}
__device__ __forceinline__ float dpp_sum16(float v) {
  v += __builtin_bit_cast(float, __builtin_amdgcn_update_dpp(
      0, __builtin_bit_cast(int, v), 0xB1, 0xF, 0xF, true));
  v += __builtin_bit_cast(float, __builtin_amdgcn_update_dpp(
      0, __builtin_bit_cast(int, v), 0x4E, 0xF, 0xF, true));
  v += __builtin_bit_cast(float, __builtin_amdgcn_update_dpp(
      0, __builtin_bit_cast(int, v), 0x141, 0xF, 0xF, true));
  v += __builtin_bit_cast(float, __builtin_amdgcn_update_dpp(
      0, __builtin_bit_cast(int, v), 0x140, 0xF, 0xF, true));
  return v;
}

__device__ __forceinline__ float gelu_fast(float v) {
  const float t = 0.7978845608028654f * (v + 0.044715f * v * v * v);
  const float e = __expf(2.0f * t);
  return v * e * __builtin_amdgcn_rcpf(e + 1.0f);
}

// ------------------------------------------------ weight convert+transpose
// W [K][N] fp32 -> Wt [N][K] bf16 (batched over blockIdx.z experts)
__global__ __launch_bounds__(256) void wconv_kernel(const float* __restrict__ W,
                                                    unsigned short* __restrict__ Wt,
                                                    int K, int N) {
  __shared__ float tile[64][65];
  const size_t eo = (size_t)blockIdx.z * K * N;
  const int n0 = blockIdx.x * 64, k0 = blockIdx.y * 64;
  for (int i = threadIdx.x; i < 64 * 64; i += 256) {
    const int r = i >> 6, c = i & 63;           // r: k, c: n (coalesced read)
    tile[r][c] = W[eo + (size_t)(k0 + r) * N + n0 + c];
  }
  __syncthreads();
  for (int i = threadIdx.x; i < 64 * 64; i += 256) {
    const int r = i >> 6, c = i & 63;           // r: n, c: k (coalesced write)
    Wt[eo + (size_t)(n0 + r) * K + k0 + c] = f2bf(tile[c][r]);
  }
}

// ---------------------------------------------------------------- LayerNorm -> bf16
__global__ void ln_kernel(const float* __restrict__ x,
                          const float* __restrict__ g,
                          const float* __restrict__ bta,
                          unsigned short* __restrict__ h) {
  const int t = blockIdx.x;
  const float* xr = x + (size_t)t * Dq;
  float s = 0.f, s2 = 0.f;
  for (int d = threadIdx.x; d < Dq; d += blockDim.x) {
    float v = xr[d]; s += v; s2 += v * v;
  }
#pragma unroll
  for (int m = 16; m; m >>= 1) { s += __shfl_xor(s, m, 32); s2 += __shfl_xor(s2, m, 32); }
  __shared__ float ssum[8], ssum2[8];
  const int wid = threadIdx.x >> 5, lane = threadIdx.x & 31;
  if (lane == 0) { ssum[wid] = s; ssum2[wid] = s2; }
  __syncthreads();
  if (wid == 0) {
    s  = (lane < 8) ? ssum[lane]  : 0.f;
    s2 = (lane < 8) ? ssum2[lane] : 0.f;
#pragma unroll
    for (int m = 1; m < 8; m <<= 1) { s += __shfl_xor(s, m, 32); s2 += __shfl_xor(s2, m, 32); }
    if (lane == 0) { ssum[0] = s; ssum2[0] = s2; }
  }
  __syncthreads();
  const float mean = ssum[0] / (float)Dq;
  const float var  = ssum2[0] / (float)Dq - mean * mean;
  const float inv  = rsqrtf(var + 1e-5f);
  unsigned short* hr = h + (size_t)t * Dq;
  for (int d = threadIdx.x; d < Dq; d += blockDim.x)
    hr[d] = f2bf((xr[d] - mean) * inv * g[d] + bta[d]);
}

// ---------------------------------------------------------------- WMMA GEMM
// C[M,N] = A[M,K] x Bt[N,K]^T.  A: bf16 row-major, Bt: bf16 pre-transposed.
// Block 128x128, 8 waves (2Mx4N), K-step 64, async global->LDS staging.
// EPI 0: fp32 C.   EPI 1: ragged, +bias, GELU -> bf16.
// EPI 2: ragged, +bias, weighted atomic scatter.  EPI 3: bf16 C * cscale.
struct GemmParams {
  const unsigned short* Abf;
  const unsigned short* Bt;      // [(E x) N x K] bf16
  float*                C;
  unsigned short*       Cbf;
  const float*          bias;
  const int*            seg_off;
  const int*            seg_cnt;
  const int*            tok_list;
  const float*          row_w;
  float*                out;
  float                 cscale;
  int M, N, K;
};

template <int EPI>
__global__ __launch_bounds__(256) void gemm128(GemmParams p) {
  __shared__ alignas(16) unsigned short sA[128][72];   // [m][k], pad 8
  __shared__ alignas(16) unsigned short sB[128][72];   // [n][k], pad 8

  const int tid  = threadIdx.x;
  const int lane = tid & 31, wid = tid >> 5;
  const int waveM = wid >> 2, waveN = wid & 3;
  const int half = lane >> 4, l16 = lane & 15;

  int e = 0, rowBase = 0, rowCnt = p.M;
  if (EPI == 1 || EPI == 2) { e = blockIdx.z; rowBase = p.seg_off[e]; rowCnt = p.seg_cnt[e]; }
  const int m0 = blockIdx.x * 128;
  if (m0 >= rowCnt) return;
  const int n0 = blockIdx.y * 128;
  const bool fullM = (EPI == 0 || EPI == 3) || (m0 + 128 <= rowCnt);
  const unsigned short* Bt = p.Bt + (size_t)e * p.K * p.N;

  const v8f vzero = {};
  v8f acc[4][2];
#pragma unroll
  for (int i = 0; i < 4; i++)
#pragma unroll
    for (int j = 0; j < 2; j++) acc[i][j] = vzero;

  for (int k0 = 0; k0 < p.K; k0 += 64) {
    // ---- stage A: 128 rows x 64 k, 16B async copies
#pragma unroll
    for (int i = 0; i < 4; i++) {
      const int idx = tid + i * 256;            // 0..1023
      const int r = idx >> 3, kv = idx & 7;
      if (fullM || (m0 + r) < rowCnt)
        copy16(&sA[r][kv * 8],
               p.Abf + (size_t)(rowBase + m0 + r) * p.K + k0 + kv * 8);
      else {
        const uint4 z = {0u, 0u, 0u, 0u};
        *(uint4*)&sA[r][kv * 8] = z;
      }
    }
    // ---- stage B: 128 n-rows x 64 k, 16B async copies (pre-transposed)
#pragma unroll
    for (int i = 0; i < 4; i++) {
      const int idx = tid + i * 256;            // 0..1023
      const int n = idx >> 3, kv = idx & 7;
      copy16(&sB[n][kv * 8], Bt + (size_t)(n0 + n) * p.K + k0 + kv * 8);
    }
    wait_async();
    __syncthreads();

#pragma unroll
    for (int ks = 0; ks < 2; ks++) {
      const int kb = ks * 32;
      v16bf af[4];
#pragma unroll
      for (int i = 0; i < 4; i++) {
        const int r = waveM * 64 + i * 16 + l16;
        const int klo = kb + half * 8;
        af[i] = frag_from(&sA[r][klo], &sA[r][klo + 16]);
      }
      v16bf bfm[2];
#pragma unroll
      for (int j = 0; j < 2; j++) {
        const int n = waveN * 32 + j * 16 + l16;
        const int klo = kb + half * 8;
        bfm[j] = frag_from(&sB[n][klo], &sB[n][klo + 16]);
      }
#pragma unroll
      for (int i = 0; i < 4; i++)
#pragma unroll
        for (int j = 0; j < 2; j++)
          acc[i][j] = wmma_bf16(af[i], bfm[j], acc[i][j]);
    }
    __syncthreads();
  }

  // ---- epilogue
#pragma unroll
  for (int i = 0; i < 4; i++) {
#pragma unroll
    for (int j = 0; j < 2; j++) {
      const int col = n0 + waveN * 32 + j * 16 + l16;
#pragma unroll
      for (int r = 0; r < 8; r++) {
        const int row = m0 + waveM * 64 + i * 16 + r + half * 8;
        if (!fullM && row >= rowCnt) continue;
        float v = acc[i][j][r];
        if (EPI == 0) {
          p.C[(size_t)row * p.N + col] = v;
        } else if (EPI == 3) {
          p.Cbf[(size_t)row * p.N + col] = f2bf(v * p.cscale);
        } else if (EPI == 1) {
          v += p.bias[(size_t)e * p.N + col];
          p.Cbf[(size_t)(rowBase + row) * p.N + col] = f2bf(gelu_fast(v));
        } else {
          v += p.bias[(size_t)e * p.N + col];
          const int grow = rowBase + row;
          atomicAdd(&p.out[(size_t)p.tok_list[grow] * p.N + col],
                    p.row_w[grow] * v);
        }
      }
    }
  }
}

// ---------------------------------------------------------------- Flash attention
// bf16 Q (pre-scaled by 1/sqrt(dh)) / K / V in, bf16 O out. 64-key tiles,
// async Q/K staging, DPP16 online softmax, 16 WMMAs per tile.
__global__ __launch_bounds__(128) void flash_attn(const unsigned short* __restrict__ qb,
                                                  const unsigned short* __restrict__ kb,
                                                  const unsigned short* __restrict__ vb,
                                                  unsigned short* __restrict__ ob) {
  __shared__ alignas(16) unsigned short sQ[64][72];    // [q][d]
  __shared__ alignas(16) unsigned short sK[64][72];    // [key][d]
  __shared__ alignas(16) unsigned short sV[64][72];    // transposed [d][key]
  __shared__ alignas(16) unsigned short sP[4][16][72]; // per-wave P tile

  const int tid  = threadIdx.x;
  const int lane = tid & 31, wid = tid >> 5;
  const int half = lane >> 4, l16 = lane & 15;
  const int qbase = blockIdx.x * 64;
  const size_t base = (size_t)blockIdx.z * Sq * Dq + (size_t)blockIdx.y * DH;

  // stage Q (64 x 64 bf16) once: pure async row copies
#pragma unroll
  for (int i = 0; i < 4; i++) {
    const int idx = tid + i * 128;              // 0..511
    const int r = idx >> 3, kv = idx & 7;
    copy16(&sQ[r][kv * 8], qb + base + (size_t)(qbase + r) * Dq + kv * 8);
  }
  wait_async();
  __syncthreads();
  v16bf aq[2];
#pragma unroll
  for (int ks = 0; ks < 2; ks++) {
    const int r = wid * 16 + l16;
    const int klo = ks * 32 + half * 8;
    aq[ks] = frag_from(&sQ[r][klo], &sQ[r][klo + 16]);
  }

  const v8f vzero = {};
  float mrow[8], lrow[8], alpha[8];
  v8f o[4];
#pragma unroll
  for (int r = 0; r < 8; r++) { mrow[r] = -1e30f; lrow[r] = 0.f; }
#pragma unroll
  for (int n = 0; n < 4; n++) o[n] = vzero;

  for (int kt = 0; kt < Sq / 64; kt++) {
    const int kbase = kt * 64;
    // K: async row copies; V: bf16 load + transposed b16 stores
#pragma unroll
    for (int i = 0; i < 4; i++) {
      const int idx = tid + i * 128;            // 0..511
      const int r = idx >> 3, kv = idx & 7;
      copy16(&sK[r][kv * 8], kb + base + (size_t)(kbase + r) * Dq + kv * 8);
    }
#pragma unroll
    for (int i = 0; i < 8; i++) {
      const int idx = tid + i * 128;            // 0..1023
      const int r = idx >> 4, dq = idx & 15;
      const uint2 v2 = *(const uint2*)(vb + base + (size_t)(kbase + r) * Dq + dq * 4);
      const int d = dq * 4;
      sV[d + 0][r] = (unsigned short)(v2.x);
      sV[d + 1][r] = (unsigned short)(v2.x >> 16);
      sV[d + 2][r] = (unsigned short)(v2.y);
      sV[d + 3][r] = (unsigned short)(v2.y >> 16);
    }
    wait_async();
    __syncthreads();

    // scores: 16 q x 64 keys as four 16x16 tiles
    v8f c[4];
#pragma unroll
    for (int j = 0; j < 4; j++) c[j] = vzero;
#pragma unroll
    for (int ks = 0; ks < 2; ks++) {
      const int klo = ks * 32 + half * 8;
#pragma unroll
      for (int j = 0; j < 4; j++) {
        v16bf bk = frag_from(&sK[j * 16 + l16][klo], &sK[j * 16 + l16][klo + 16]);
        c[j] = wmma_bf16(aq[ks], bk, c[j]);
      }
    }

    // online softmax per row (row = r + half*8)
#pragma unroll
    for (int r = 0; r < 8; r++) {
      float tm = fmaxf(fmaxf(c[0][r], c[1][r]), fmaxf(c[2][r], c[3][r]));
      tm = dpp_max16(tm);
      const float nm = fmaxf(mrow[r], tm);
      const float al = __expf(mrow[r] - nm);
      float pj[4];
#pragma unroll
      for (int j = 0; j < 4; j++) pj[j] = __expf(c[j][r] - nm);
      float rs = (pj[0] + pj[1]) + (pj[2] + pj[3]);
      rs = dpp_sum16(rs);
      lrow[r] = lrow[r] * al + rs;
      mrow[r] = nm; alpha[r] = al;
      const int row = r + half * 8;
#pragma unroll
      for (int j = 0; j < 4; j++) sP[wid][row][j * 16 + l16] = f2bf(pj[j]);
    }
#pragma unroll
    for (int n = 0; n < 4; n++)
#pragma unroll
      for (int r = 0; r < 8; r++) o[n][r] = o[n][r] * alpha[r];

    // O += P @ V
#pragma unroll
    for (int ks = 0; ks < 2; ks++) {
      const int klo = ks * 32 + half * 8;
      v16bf ap = frag_from(&sP[wid][l16][klo], &sP[wid][l16][klo + 16]);
      const int kk = ks * 32 + half * 8;
#pragma unroll
      for (int n = 0; n < 4; n++) {
        v16bf bv = frag_from(&sV[n * 16 + l16][kk], &sV[n * 16 + l16][kk + 16]);
        o[n] = wmma_bf16(ap, bv, o[n]);
      }
    }
    __syncthreads();
  }

  float linv[8];
#pragma unroll
  for (int r = 0; r < 8; r++) linv[r] = __builtin_amdgcn_rcpf(lrow[r]);
#pragma unroll
  for (int n = 0; n < 4; n++) {
    const int d = n * 16 + l16;
#pragma unroll
    for (int r = 0; r < 8; r++) {
      const int row = qbase + wid * 16 + r + half * 8;
      ob[base + (size_t)row * Dq + d] = f2bf(o[n][r] * linv[r]);
    }
  }
}

// ---------------------------------------------------------------- residual
__global__ void resid_kernel(const float* __restrict__ x,
                             const float* __restrict__ pb,
                             float* __restrict__ x1,
                             float* __restrict__ out) {
  const size_t i = (size_t)blockIdx.x * blockDim.x + threadIdx.x;
  if (i < (size_t)NT * Dq) { const float v = x[i] + pb[i]; x1[i] = v; out[i] = v; }
}

// ---------------------------------------------------------------- MoE routing
__global__ void zero8(int* a, int* b2) {
  if (threadIdx.x < 8) { a[threadIdx.x] = 0; b2[threadIdx.x] = 0; }
}

__global__ void gate_kernel(const float* __restrict__ x1,
                            const float* __restrict__ Wg,
                            float* __restrict__ topw, int* __restrict__ topi,
                            int* __restrict__ cnt) {
  const int t = blockIdx.x;
  const int tid = threadIdx.x, wid = tid >> 5, lane = tid & 31;
  __shared__ float logit[8];
  const float* xr = x1 + (size_t)t * Dq;
  float s = 0.f;
  for (int d = lane; d < Dq; d += 32) s += xr[d] * Wg[(size_t)d * Eq + wid];
#pragma unroll
  for (int m = 16; m; m >>= 1) s += __shfl_xor(s, m, 32);
  if (lane == 0) logit[wid] = s;
  __syncthreads();
  if (tid == 0) {
    float mx = -1e30f;
#pragma unroll
    for (int e2 = 0; e2 < 8; e2++) mx = fmaxf(mx, logit[e2]);
    float pe[8];
#pragma unroll
    for (int e2 = 0; e2 < 8; e2++) pe[e2] = __expf(logit[e2] - mx);
    int i0 = 0; float v0 = -1.f;
#pragma unroll
    for (int e2 = 0; e2 < 8; e2++) if (pe[e2] > v0) { v0 = pe[e2]; i0 = e2; }
    int i1 = 0; float v1 = -1.f;
#pragma unroll
    for (int e2 = 0; e2 < 8; e2++) if (e2 != i0 && pe[e2] > v1) { v1 = pe[e2]; i1 = e2; }
    const float inv = 1.0f / (v0 + v1);
    topi[t * 2] = i0; topi[t * 2 + 1] = i1;
    topw[t * 2] = v0 * inv; topw[t * 2 + 1] = v1 * inv;
    atomicAdd(&cnt[i0], 1); atomicAdd(&cnt[i1], 1);
  }
}

__global__ void scan_kernel(const int* __restrict__ cnt, int* __restrict__ offs,
                            int* __restrict__ cursor) {
  if (threadIdx.x == 0) {
    int a = 0;
#pragma unroll
    for (int e2 = 0; e2 < 8; e2++) { offs[e2] = a; a += cnt[e2]; cursor[e2] = 0; }
  }
}

__global__ void assign_kernel(const int* __restrict__ topi,
                              const float* __restrict__ topw,
                              const int* __restrict__ offs, int* __restrict__ cursor,
                              int* __restrict__ tokl, float* __restrict__ roww) {
  const int t = blockIdx.x * blockDim.x + threadIdx.x;
  if (t >= NT) return;
#pragma unroll
  for (int j = 0; j < 2; j++) {
    const int e2 = topi[t * 2 + j];
    const int pos = atomicAdd(&cursor[e2], 1);
    const int row = offs[e2] + pos;
    tokl[row] = t; roww[row] = topw[t * 2 + j];
  }
}

__global__ void gather_kernel(const float* __restrict__ x1,
                              const int* __restrict__ tokl,
                              unsigned short* __restrict__ Xe) {
  const int row = blockIdx.x;
  const int t = tokl[row];
  const float4* xr = (const float4*)(x1 + (size_t)t * Dq);
  unsigned short* o = Xe + (size_t)row * Dq;
  for (int dv = threadIdx.x; dv < Dq / 4; dv += blockDim.x) {
    const float4 v4 = xr[dv];
    uint2 pk; pk.x = pack2(v4.x, v4.y); pk.y = pack2(v4.z, v4.w);
    *(uint2*)&o[dv * 4] = pk;
  }
}

// ---------------------------------------------------------------- launcher
extern "C" void kernel_launch(void* const* d_in, const int* in_sizes, int n_in,
                              void* d_out, int out_size, void* d_ws, size_t ws_size,
                              hipStream_t stream) {
  (void)in_sizes; (void)n_in; (void)out_size; (void)ws_size;
  const float* x    = (const float*)d_in[0];
  const float* ln_g = (const float*)d_in[1];
  const float* ln_b = (const float*)d_in[2];
  const float* Wq   = (const float*)d_in[3];
  const float* Wk   = (const float*)d_in[4];
  const float* Wv   = (const float*)d_in[5];
  const float* Wo   = (const float*)d_in[6];
  const float* Wg   = (const float*)d_in[7];
  const float* W1   = (const float*)d_in[8];
  const float* b1   = (const float*)d_in[9];
  const float* W2   = (const float*)d_in[10];
  const float* b2   = (const float*)d_in[11];
  float* out = (float*)d_out;

  char* ws = (char*)d_ws;
  size_t off = 0;
  auto alloc = [&](size_t bytes) -> void* {
    void* p = ws + off;
    off = (off + bytes + 255) & ~(size_t)255;
    return p;
  };
  typedef unsigned short u16;
  u16*   hb   = (u16*)alloc((size_t)NT * Dq * 2);
  u16*   qb   = (u16*)alloc((size_t)NT * Dq * 2);
  u16*   kbuf = (u16*)alloc((size_t)NT * Dq * 2);
  u16*   vbuf = (u16*)alloc((size_t)NT * Dq * 2);
  u16*   obuf = (u16*)alloc((size_t)NT * Dq * 2);
  float* pb   = (float*)alloc((size_t)NT * Dq * 4);
  float* x1   = (float*)alloc((size_t)NT * Dq * 4);
  float* topw = (float*)alloc((size_t)NT * 2 * 4);
  int*   topi = (int*)alloc((size_t)NT * 2 * 4);
  int*   cnt  = (int*)alloc(64);
  int*   offs = (int*)alloc(64);
  int*   curs = (int*)alloc(64);
  int*   tokl = (int*)alloc((size_t)NA * 4);
  float* roww = (float*)alloc((size_t)NA * 4);
  u16*   Xe   = (u16*)alloc((size_t)NA * Dq * 2);
  u16*   H1   = (u16*)alloc((size_t)NA * Fq * 2);
  u16*   Wqt  = (u16*)alloc((size_t)Dq * Dq * 2);
  u16*   Wkt  = (u16*)alloc((size_t)Dq * Dq * 2);
  u16*   Wvt  = (u16*)alloc((size_t)Dq * Dq * 2);
  u16*   Wot  = (u16*)alloc((size_t)Dq * Dq * 2);
  u16*   W1t  = (u16*)alloc((size_t)Eq * Dq * Fq * 2);
  u16*   W2t  = (u16*)alloc((size_t)Eq * Fq * Dq * 2);

  // 0) weight convert + transpose to bf16 [N][K]
  const dim3 gWd(Dq / 64, Dq / 64, 1);
  wconv_kernel<<<gWd, 256, 0, stream>>>(Wq, Wqt, Dq, Dq);
  wconv_kernel<<<gWd, 256, 0, stream>>>(Wk, Wkt, Dq, Dq);
  wconv_kernel<<<gWd, 256, 0, stream>>>(Wv, Wvt, Dq, Dq);
  wconv_kernel<<<gWd, 256, 0, stream>>>(Wo, Wot, Dq, Dq);
  wconv_kernel<<<dim3(Fq / 64, Dq / 64, Eq), 256, 0, stream>>>(W1, W1t, Dq, Fq);
  wconv_kernel<<<dim3(Dq / 64, Fq / 64, Eq), 256, 0, stream>>>(W2, W2t, Fq, Dq);

  // 1) pre-norm -> bf16
  ln_kernel<<<NT, 256, 0, stream>>>(x, ln_g, ln_b, hb);

  // 2) Q/K/V projections -> bf16 (1/sqrt(dh) folded into Q)
  const dim3 gProj(NT / 128, Dq / 128, 1);
  GemmParams pg{};
  pg.M = NT; pg.N = Dq; pg.K = Dq; pg.Abf = hb;
  pg.cscale = 0.125f; pg.Bt = Wqt; pg.Cbf = qb;
  gemm128<3><<<gProj, 256, 0, stream>>>(pg);
  pg.cscale = 1.0f;   pg.Bt = Wkt; pg.Cbf = kbuf;
  gemm128<3><<<gProj, 256, 0, stream>>>(pg);
  pg.Bt = Wvt; pg.Cbf = vbuf;
  gemm128<3><<<gProj, 256, 0, stream>>>(pg);

  // 3) flash attention (bf16 in/out)
  flash_attn<<<dim3(Sq / 64, Hh, Bq), 128, 0, stream>>>(qb, kbuf, vbuf, obuf);

  // 4) output projection (fp32 out) + residual
  GemmParams po{};
  po.M = NT; po.N = Dq; po.K = Dq; po.Abf = obuf; po.Bt = Wot; po.C = pb;
  gemm128<0><<<gProj, 256, 0, stream>>>(po);
  resid_kernel<<<(NT * Dq) / 256, 256, 0, stream>>>(x, pb, x1, out);

  // 5) MoE routing
  zero8<<<1, 32, 0, stream>>>(cnt, curs);
  gate_kernel<<<NT, 256, 0, stream>>>(x1, Wg, topw, topi, cnt);
  scan_kernel<<<1, 32, 0, stream>>>(cnt, offs, curs);
  assign_kernel<<<NT / 256, 256, 0, stream>>>(topi, topw, offs, curs, tokl, roww);
  gather_kernel<<<NA, 256, 0, stream>>>(x1, tokl, Xe);

  // 6) MoE GEMM1: H1 = gelu(Xe @ W1[e] + b1[e])
  GemmParams p1{};
  p1.M = NT; p1.N = Fq; p1.K = Dq;
  p1.Abf = Xe; p1.Bt = W1t; p1.bias = b1; p1.Cbf = H1;
  p1.seg_off = offs; p1.seg_cnt = cnt;
  gemm128<1><<<dim3(NT / 128, Fq / 128, Eq), 256, 0, stream>>>(p1);

  // 7) MoE GEMM2: out[tok] += w * (H1 @ W2[e] + b2[e])
  GemmParams p2{};
  p2.M = NT; p2.N = Dq; p2.K = Fq;
  p2.Abf = H1; p2.Bt = W2t; p2.bias = b2; p2.out = out;
  p2.seg_off = offs; p2.seg_cnt = cnt; p2.tok_list = tokl; p2.row_w = roww;
  gemm128<2><<<dim3(NT / 128, Dq / 128, Eq), 256, 0, stream>>>(p2);
}
